// UNetBlock1D_23613730194015
// MI455X (gfx1250) — matrix operand order repaired and checked by reading, not compiled, not run on hardware
//
#include <hip/hip_runtime.h>

// ---------------------------------------------------------------------------
// UNetBlock1D for MI455X (gfx1250): bf16 WMMA GEMMs + fp32 norms/softmax.
// B=16, CIN=256, COUT=512, L=1024, K=3, HEADS=8, ch=64, GROUPS=32
// K=3 convs are decomposed into 3 unshifted GEMM passes with an output
// column shift in the epilogue, so the inner loop has zero branches/masks.
// ---------------------------------------------------------------------------

typedef __attribute__((ext_vector_type(16))) __bf16 v16bf;
typedef __attribute__((ext_vector_type(8)))  float  v8f;

#define TM 128
#define TN 64
#define TK 32

#define C_B     16
#define C_CIN   256
#define C_COUT  512
#define C_L     1024
#define C_HEADS 8
#define C_CH    64
#define C_BH    (C_B * C_HEADS)
#define C_EPS   1e-5f

union Frag16 { v16bf v; uint4 q[2]; };
union Half8  { __bf16 e[8]; uint4 q; };

// ---------------------------------------------------------------------------
// Tiled WMMA GEMM (one conv tap / one plain GEMM per launch):
//   out[bb][m][n + colShift] (+)= sum_kc A[m][kc] * X[bb][kc][n]  (+bias/res)
// A: row stride aRowStride elems (slice of packed weights, or per-batch via
//    aBatch). X: [Kc][Ln] per batch, row-major. Out: [M][Ln] per batch.
// flags: bit0 = bf16 output (outB), bit1 = accumulate onto existing outF.
// Block 256 threads (8 waves); tile 128x64; K-step 32 (bf16 WMMA).
// B tile is stored TRANSPOSED in LDS ([n][k]) so both A and B fragments are
// two contiguous 16-byte reads per lane (k = 8h.., 16+8h..; ds_load_b128 pairs).
// ---------------------------------------------------------------------------
__global__ __launch_bounds__(256) void wmma_gemm(
    const __bf16* __restrict__ A, const __bf16* __restrict__ Xin,
    const float* __restrict__ bias, const float* __restrict__ res,
    float* __restrict__ outF, __bf16* __restrict__ outB,
    int M, int Kc, int Ln, int aRowStride,
    long aBatch, long xBatch, long oBatch,
    int colShift, int flags)
{
    __shared__ __bf16 ldsA [TM * TK];   // [row][k]
    __shared__ __bf16 ldsBt[TN * TK];   // transposed: [n][k]

    const int tid  = threadIdx.x;
    const int wv   = tid >> 5;          // wave 0..7 -> 16-row M strip
    const int lane = tid & 31;
    const int lo   = lane & 15;
    const int hi   = lane >> 4;
    const int l0   = blockIdx.x * TN;
    const int m0   = blockIdx.y * TM;
    const int bb   = blockIdx.z;

    const __bf16* Ab = A   + (size_t)bb * (size_t)aBatch;
    const __bf16* Xb = Xin + (size_t)bb * (size_t)xBatch;

    // A-stage mapping: thread -> (row, 16-col half)
    const int arow  = tid >> 1;
    const int ahalf = (tid & 1) * 16;
    // B-stage mapping: thread -> (col n, 8-row k segment)
    const int bn    = tid & 63;
    const int bkseg = (tid >> 6) * 8;

    v8f acc[4];
    #pragma unroll
    for (int i = 0; i < 4; ++i)
        #pragma unroll
        for (int e = 0; e < 8; ++e) acc[i][e] = 0.0f;

    for (int kc0 = 0; kc0 < Kc; kc0 += TK) {
        // ---- stage A tile (TM x TK): 2 x uint4 per thread, aligned ----
        {
            uint4 q0 = {0, 0, 0, 0}, q1 = {0, 0, 0, 0};
            const int m = m0 + arow;
            if (m < M) {
                const uint4* src = (const uint4*)(
                    Ab + (size_t)m * aRowStride + kc0 + ahalf);
                q0 = src[0];
                q1 = src[1];
            }
            uint4* dst = (uint4*)&ldsA[arow * TK + ahalf];
            dst[0] = q0;
            dst[1] = q1;
        }
        // ---- stage B tile transposed: gather 8 rows, one b128 LDS store ----
        {
            const __bf16* src = Xb + (size_t)(kc0 + bkseg) * Ln + l0 + bn;
            Half8 t;
            #pragma unroll
            for (int j = 0; j < 8; ++j) t.e[j] = src[(size_t)j * Ln];
            *(uint4*)&ldsBt[bn * TK + bkseg] = t.q;
        }
        __syncthreads();

        // ---- A fragment: two contiguous 16B chunks per lane ----
        Frag16 af;
        {
            const int mrow = wv * 16 + lo;
            af.q[0] = *(const uint4*)&ldsA[mrow * TK + hi * 8];
            af.q[1] = *(const uint4*)&ldsA[mrow * TK + 16 + hi * 8];
        }
        // ---- 4 N-subtiles: contiguous B fragment + WMMA ----
        #pragma unroll
        for (int ns = 0; ns < 4; ++ns) {
            Frag16 bf;
            const uint4* bp =
                (const uint4*)&ldsBt[(ns * 16 + lo) * TK + hi * 8];
            bf.q[0] = bp[0];
            bf.q[1] = bp[2];
            acc[ns] = __builtin_amdgcn_wmma_f32_16x16x32_bf16(
                false, af.v, false, bf.v, (short)0, acc[ns], false, false);
        }
        __syncthreads();
    }

    // ---- epilogue: C layout (M = wv*16 + hi*8 + r, N = lo), shifted col ----
    #pragma unroll
    for (int ns = 0; ns < 4; ++ns) {
        const int col = l0 + ns * 16 + lo + colShift;
        if (col < 0 || col >= Ln) continue;
        #pragma unroll
        for (int r = 0; r < 8; ++r) {
            const int m = m0 + wv * 16 + hi * 8 + r;
            if (m < M) {
                float v = acc[ns][r];
                if (bias) v += bias[m];
                const size_t oidx =
                    (size_t)bb * (size_t)oBatch + (size_t)m * Ln + col;
                if (flags & 2) v += outF[oidx];
                if (res)       v += res[oidx];
                if (flags & 1) outB[oidx] = (__bf16)v;
                else           outF[oidx] = v;
            }
        }
    }
}

// ---------------------------------------------------------------------------
// GroupNorm statistics: one block per (batch, group); fp32 reductions.
// ---------------------------------------------------------------------------
__global__ __launch_bounds__(256) void gn_stats(
    const float* __restrict__ X, float* __restrict__ st,
    int C, int Lx, int cPerG, float eps)
{
    const int gid    = blockIdx.x;
    const int groups = C / cPerG;
    const int b = gid / groups, g = gid % groups;
    const float* base = X + ((size_t)b * C + (size_t)g * cPerG) * Lx;
    const int n = cPerG * Lx;

    float s = 0.0f, ss = 0.0f;
    for (int i = threadIdx.x; i < n; i += 256) {
        const float v = base[i];
        s += v; ss += v * v;
    }
    __shared__ float shs[256], shq[256];
    shs[threadIdx.x] = s; shq[threadIdx.x] = ss;
    __syncthreads();
    for (int off = 128; off > 0; off >>= 1) {
        if (threadIdx.x < off) {
            shs[threadIdx.x] += shs[threadIdx.x + off];
            shq[threadIdx.x] += shq[threadIdx.x + off];
        }
        __syncthreads();
    }
    if (threadIdx.x == 0) {
        const float mu  = shs[0] / (float)n;
        const float var = shq[0] / (float)n - mu * mu;
        st[gid * 2]     = mu;
        st[gid * 2 + 1] = rsqrtf(var + eps);
    }
}

// ---------------------------------------------------------------------------
// GroupNorm apply (+ optional SiLU), fp32 in -> bf16 out.
// ---------------------------------------------------------------------------
__global__ void gn_apply(const float* __restrict__ X, const float* __restrict__ st,
                         const float* __restrict__ w, const float* __restrict__ b,
                         __bf16* __restrict__ out,
                         int C, int Lx, int cPerG, int groups, int silu, int total)
{
    const int idx = blockIdx.x * blockDim.x + threadIdx.x;
    if (idx >= total) return;
    const int c  = (idx / Lx) % C;
    const int bb = idx / (Lx * C);
    const int gid = bb * groups + (c / cPerG);
    const float mu = st[gid * 2], ri = st[gid * 2 + 1];
    float v = (X[idx] - mu) * ri * w[c] + b[c];
    if (silu) v = v * (1.0f / (1.0f + __expf(-v)));
    out[idx] = (__bf16)v;
}

// ---------------------------------------------------------------------------
// Pack conv weights: fp32 [M][Cin][Kt] -> bf16 [M][Kt][Cin]
// (Cin==1, Kt==1 degenerates to plain fp32->bf16 copy.)
// ---------------------------------------------------------------------------
__global__ void pack_w(const float* __restrict__ W, __bf16* __restrict__ out,
                       int M, int Cin, int Kt, int total)
{
    const int idx = blockIdx.x * blockDim.x + threadIdx.x;
    if (idx >= total) return;
    const int k  = idx % Kt;
    const int ci = (idx / Kt) % Cin;
    const int m  = idx / (Kt * Cin);
    out[((size_t)m * Kt + k) * Cin + ci] = (__bf16)W[idx];
}

// ---------------------------------------------------------------------------
// Split qkv [B][1536][L] bf16 into Qt[bh][l][c] (scaled by 1/sqrt(ch)),
// Kb[bh][c][l], Vb[bh][c][l].  co = h*192 + c*3 + s (reference reshape).
// ---------------------------------------------------------------------------
__global__ void qkv_split(const __bf16* __restrict__ QKV,
                          __bf16* __restrict__ Qt, __bf16* __restrict__ Kb,
                          __bf16* __restrict__ Vb, int total)
{
    const int idx = blockIdx.x * blockDim.x + threadIdx.x;
    if (idx >= total) return;
    const int l  = idx & (C_L - 1);
    const int co = (idx >> 10) % (3 * C_COUT);
    const int b  = idx / (3 * C_COUT * C_L);
    const int h   = co / (3 * C_CH);
    const int rem = co - h * (3 * C_CH);
    const int c   = rem / 3;
    const int s   = rem - c * 3;
    const size_t base = (size_t)(b * C_HEADS + h) * (size_t)(C_CH * C_L);
    const float v = (float)QKV[idx];
    if (s == 0)      Qt[base + (size_t)l * C_CH + c] = (__bf16)(v * 0.125f);
    else if (s == 1) Kb[base + (size_t)c * C_L + l] = (__bf16)v;
    else             Vb[base + (size_t)c * C_L + l] = (__bf16)v;
}

// ---------------------------------------------------------------------------
// Row softmax over fp32 scores S[row][0..1023], write transposed bf16
// Pt[bh][j][i] so the PV GEMM consumes a row-major B matrix.
// ---------------------------------------------------------------------------
__global__ __launch_bounds__(256) void softmax_pt(
    const float* __restrict__ S, __bf16* __restrict__ Pt)
{
    const int row = blockIdx.x;          // bh_local * L + i
    const int bh  = row / C_L;
    const int i   = row % C_L;
    const float* src = S + (size_t)row * C_L;

    float vals[4];
    float mx = -3.4e38f;
    #pragma unroll
    for (int t = 0; t < 4; ++t) {
        vals[t] = src[threadIdx.x + t * 256];
        mx = fmaxf(mx, vals[t]);
    }
    __shared__ float red[256];
    red[threadIdx.x] = mx;
    __syncthreads();
    for (int off = 128; off > 0; off >>= 1) {
        if (threadIdx.x < off)
            red[threadIdx.x] = fmaxf(red[threadIdx.x], red[threadIdx.x + off]);
        __syncthreads();
    }
    mx = red[0];
    __syncthreads();

    float sum = 0.0f;
    #pragma unroll
    for (int t = 0; t < 4; ++t) {
        vals[t] = __expf(vals[t] - mx);
        sum += vals[t];
    }
    red[threadIdx.x] = sum;
    __syncthreads();
    for (int off = 128; off > 0; off >>= 1) {
        if (threadIdx.x < off)
            red[threadIdx.x] += red[threadIdx.x + off];
        __syncthreads();
    }
    const float inv = 1.0f / red[0];

    __bf16* dst = Pt + (size_t)bh * C_L * C_L + i;
    #pragma unroll
    for (int t = 0; t < 4; ++t) {
        const int j = threadIdx.x + t * 256;
        dst[(size_t)j * C_L] = (__bf16)(vals[t] * inv);
    }
}

// ---------------------------------------------------------------------------
// Host side
// ---------------------------------------------------------------------------
extern "C" void kernel_launch(void* const* d_in, const int* in_sizes, int n_in,
                              void* d_out, int out_size, void* d_ws, size_t ws_size,
                              hipStream_t stream)
{
    (void)in_sizes; (void)n_in; (void)out_size; (void)ws_size;

    const float* x    = (const float*)d_in[0];
    const float* n0w  = (const float*)d_in[1];
    const float* n0b  = (const float*)d_in[2];
    const float* c0w  = (const float*)d_in[3];
    const float* c0b  = (const float*)d_in[4];
    const float* n1w  = (const float*)d_in[5];
    const float* n1b  = (const float*)d_in[6];
    const float* c1w  = (const float*)d_in[7];
    const float* c1b  = (const float*)d_in[8];
    const float* skw  = (const float*)d_in[9];
    const float* skb  = (const float*)d_in[10];
    const float* n2w  = (const float*)d_in[11];
    const float* n2b  = (const float*)d_in[12];
    const float* qkvw = (const float*)d_in[13];
    const float* qkvb = (const float*)d_in[14];
    const float* prw  = (const float*)d_in[15];
    const float* prb  = (const float*)d_in[16];

    // ---- workspace arena ----
    char* wsp = (char*)d_ws;
    size_t off = 0;
    auto alloc = [&](size_t bytes) -> char* {
        char* p = wsp + off;
        off += (bytes + 255) & ~(size_t)255;
        return p;
    };
    const size_t NF  = (size_t)C_B * C_COUT * C_L;
    const size_t NX  = (size_t)C_B * C_CIN * C_L;
    const size_t NHD = (size_t)C_BH * C_CH * C_L;

    float*  hbuf  = (float*) alloc(NF * 4);                 // residual trunk h
    __bf16* Xorig = (__bf16*)alloc(NX * 2);
    __bf16* X0    = (__bf16*)alloc(NX * 2);
    float*  Y0    = (float*) alloc(NF * 4);
    __bf16* H1    = (__bf16*)alloc(NF * 2);
    __bf16* H2    = (__bf16*)alloc(NF * 2);
    __bf16* QKVb  = (__bf16*)alloc((size_t)C_B * 3 * C_COUT * C_L * 2);
    __bf16* Qt    = (__bf16*)alloc(NHD * 2);
    __bf16* Kb    = (__bf16*)alloc(NHD * 2);
    __bf16* Vb    = (__bf16*)alloc(NHD * 2);
    float*  Sbuf  = (float*) alloc((size_t)16 * C_L * C_L * 4);  // 16-head chunk
    __bf16* Pt    = (__bf16*)alloc((size_t)16 * C_L * C_L * 2);
    __bf16* ATTb  = (__bf16*)alloc(NF * 2);
    __bf16* Wc0   = (__bf16*)alloc((size_t)C_COUT * C_CIN * 3 * 2);
    __bf16* Wc1   = (__bf16*)alloc((size_t)C_COUT * C_COUT * 3 * 2);
    __bf16* Wsk   = (__bf16*)alloc((size_t)C_COUT * C_CIN * 2);
    __bf16* Wq    = (__bf16*)alloc((size_t)3 * C_COUT * C_COUT * 2);
    __bf16* Wp    = (__bf16*)alloc((size_t)C_COUT * C_COUT * 2);
    float*  st0   = (float*) alloc((size_t)C_B * 32 * 2 * 4);
    float*  st1   = (float*) alloc((size_t)C_B * 32 * 2 * 4);
    float*  st2   = (float*) alloc((size_t)C_B * 32 * 2 * 4);

    const int T = 256;
    auto blocks = [](int n) { return (n + 255) / 256; };

    // ---- weight packs (fp32 -> bf16, [M][Cin][Kt] -> [M][Kt][Cin]) ----
    { int n = C_COUT * C_CIN * 3;  pack_w<<<blocks(n), T, 0, stream>>>(c0w,  Wc0, C_COUT, C_CIN, 3, n); }
    { int n = C_COUT * C_COUT * 3; pack_w<<<blocks(n), T, 0, stream>>>(c1w,  Wc1, C_COUT, C_COUT, 3, n); }
    { int n = C_COUT * C_CIN;      pack_w<<<blocks(n), T, 0, stream>>>(skw,  Wsk, C_COUT, C_CIN, 1, n); }
    { int n = 3 * C_COUT * C_COUT; pack_w<<<blocks(n), T, 0, stream>>>(qkvw, Wq,  3 * C_COUT, C_COUT, 1, n); }
    { int n = C_COUT * C_COUT;     pack_w<<<blocks(n), T, 0, stream>>>(prw,  Wp,  C_COUT, C_COUT, 1, n); }
    { int n = (int)NX;             pack_w<<<blocks(n), T, 0, stream>>>(x,    Xorig, n, 1, 1, n); } // copy

    // ---- norm0 + silu -> X0 (bf16) ----
    gn_stats<<<C_B * 32, T, 0, stream>>>(x, st0, C_CIN, C_L, C_CIN / 32, C_EPS);
    gn_apply<<<blocks((int)NX), T, 0, stream>>>(x, st0, n0w, n0b, X0,
                                                C_CIN, C_L, C_CIN / 32, 32, 1, (int)NX);

    const dim3 gConvOut(C_L / TN, C_COUT / TM, C_B);
    const long xbX = (long)C_CIN * C_L;     // X-side batch stride (CIN rows)
    const long xbF = (long)C_COUT * C_L;    // X-side batch stride (COUT rows)
    const long obF = (long)C_COUT * C_L;    // out batch stride

    // ---- conv0 (K=3): center tap first (store+bias), then shifted ACCUMs ----
    wmma_gemm<<<gConvOut, T, 0, stream>>>(Wc0 + 1 * C_CIN, X0, c0b, nullptr,
        Y0, nullptr, C_COUT, C_CIN, C_L, 3 * C_CIN, 0L, xbX, obF,  0, 0);
    wmma_gemm<<<gConvOut, T, 0, stream>>>(Wc0 + 0 * C_CIN, X0, nullptr, nullptr,
        Y0, nullptr, C_COUT, C_CIN, C_L, 3 * C_CIN, 0L, xbX, obF, +1, 2);
    wmma_gemm<<<gConvOut, T, 0, stream>>>(Wc0 + 2 * C_CIN, X0, nullptr, nullptr,
        Y0, nullptr, C_COUT, C_CIN, C_L, 3 * C_CIN, 0L, xbX, obF, -1, 2);

    // ---- norm1 + silu -> H1 ----
    gn_stats<<<C_B * 32, T, 0, stream>>>(Y0, st1, C_COUT, C_L, C_COUT / 32, C_EPS);
    gn_apply<<<blocks((int)NF), T, 0, stream>>>(Y0, st1, n1w, n1b, H1,
                                                C_COUT, C_L, C_COUT / 32, 32, 1, (int)NF);

    // ---- skip 1x1 on original x -> h (store+bias) ----
    wmma_gemm<<<gConvOut, T, 0, stream>>>(Wsk, Xorig, skb, nullptr,
        hbuf, nullptr, C_COUT, C_CIN, C_L, C_CIN, 0L, xbX, obF, 0, 0);

    // ---- conv1 (K=3) accumulates onto h ----
    wmma_gemm<<<gConvOut, T, 0, stream>>>(Wc1 + 1 * C_COUT, H1, c1b, nullptr,
        hbuf, nullptr, C_COUT, C_COUT, C_L, 3 * C_COUT, 0L, xbF, obF,  0, 2);
    wmma_gemm<<<gConvOut, T, 0, stream>>>(Wc1 + 0 * C_COUT, H1, nullptr, nullptr,
        hbuf, nullptr, C_COUT, C_COUT, C_L, 3 * C_COUT, 0L, xbF, obF, +1, 2);
    wmma_gemm<<<gConvOut, T, 0, stream>>>(Wc1 + 2 * C_COUT, H1, nullptr, nullptr,
        hbuf, nullptr, C_COUT, C_COUT, C_L, 3 * C_COUT, 0L, xbF, obF, -1, 2);

    // ---- norm2 (no silu) -> H2 ----
    gn_stats<<<C_B * 32, T, 0, stream>>>(hbuf, st2, C_COUT, C_L, C_COUT / 32, C_EPS);
    gn_apply<<<blocks((int)NF), T, 0, stream>>>(hbuf, st2, n2w, n2b, H2,
                                                C_COUT, C_L, C_COUT / 32, 32, 0, (int)NF);

    // ---- qkv 1x1: [1536] <- [512], bf16 output ----
    const dim3 gQkv(C_L / TN, (3 * C_COUT) / TM, C_B);
    wmma_gemm<<<gQkv, T, 0, stream>>>(Wq, H2, qkvb, nullptr, nullptr, QKVb,
        3 * C_COUT, C_COUT, C_L, C_COUT, 0L, xbF, (long)3 * C_COUT * C_L, 0, 1);

    // ---- split into Qt / Kb / Vb ----
    { int n = C_B * 3 * C_COUT * C_L;
      qkv_split<<<blocks(n), T, 0, stream>>>(QKVb, Qt, Kb, Vb, n); }

    // ---- attention, 8 chunks of 16 heads (S scratch = 64 MB) ----
    for (int ck = 0; ck < 8; ++ck) {
        const size_t hoff = (size_t)ck * 16 * C_CH * C_L;
        const __bf16* Qck = Qt + hoff;
        const __bf16* Kck = Kb + hoff;
        const __bf16* Vck = Vb + hoff;
        __bf16*       Ack = ATTb + hoff;   // [BH][64][L] == [B][512][L] flat

        // scores: S[i][j] = (q/8)^T k   (M=1024, N=1024, K=64)
        const dim3 gS(C_L / TN, C_L / TM, 16);
        wmma_gemm<<<gS, T, 0, stream>>>(Qck, Kck, nullptr, nullptr,
            Sbuf, nullptr, C_L, C_CH, C_L, C_CH,
            (long)C_CH * C_L, (long)C_CH * C_L, (long)C_L * C_L, 0, 0);

        // softmax rows -> transposed bf16 P^T
        softmax_pt<<<16 * C_L, T, 0, stream>>>(Sbuf, Pt);

        // out[c][i] = sum_j v[c][j] * P^T[j][i]  (M=64, N=1024, K=1024)
        const dim3 gA(C_L / TN, 1, 16);
        wmma_gemm<<<gA, T, 0, stream>>>(Vck, Pt, nullptr, nullptr,
            nullptr, Ack, C_CH, C_L, C_L, C_L,
            (long)C_CH * C_L, (long)C_L * C_L, (long)C_CH * C_L, 0, 1);
    }

    // ---- proj 1x1 + bias + residual h -> d_out (fp32) ----
    wmma_gemm<<<gConvOut, T, 0, stream>>>(Wp, ATTb, prb, hbuf,
        (float*)d_out, nullptr, C_COUT, C_COUT, C_L, C_COUT,
        0L, xbF, obF, 0, 0);
}
